// KDHR_79207786873491
// MI455X (gfx1250) — matrix operand order repaired and checked by reading, not compiled, not run on hardware
//
#include <hip/hip_runtime.h>
#include <hip/hip_bf16.h>

typedef __attribute__((ext_vector_type(16))) __bf16 v16bf;
typedef __attribute__((ext_vector_type(8)))  __bf16 v8bf;
typedef __attribute__((ext_vector_type(8)))  float  v8f;

#define NNODES 1195
#define NUSER  805
#define NITEM  390
#define DIM    64
#define BATCH  2048

static inline int ceil16(int x) { return (x + 15) & ~15; }
static inline int ceil32(int x) { return (x + 31) & ~31; }

// ---------------------------------------------------------------------------
// Operand packing: f32 -> zero-padded bf16, fully coalesced, no branches in
// the GEMM hot loop afterwards.
//   pack_a : src [rows x cols] row-major -> dst [prows x pcols] row-major
//   pack_bt: src [K x N] row-major       -> dst [Np x Kp] row-major (B^T)
// ---------------------------------------------------------------------------
__global__ void pack_a_bf16(const float* __restrict__ src, int rows, int cols,
                            __bf16* __restrict__ dst, int pcols, int n) {
  int t = blockIdx.x * blockDim.x + threadIdx.x;
  if (t >= n) return;
  int r = t / pcols, c = t - r * pcols;
  float v = (r < rows && c < cols) ? src[r * cols + c] : 0.f;
  dst[t] = (__bf16)v;
}

__global__ void pack_bt_bf16(const float* __restrict__ src, int K, int N,
                             __bf16* __restrict__ dst, int Kp, int n) {
  int t = blockIdx.x * blockDim.x + threadIdx.x;
  if (t >= n) return;
  int nn = t / Kp, k = t - nn * Kp;
  float v = (k < K && nn < N) ? src[k * N + nn] : 0.f;
  dst[t] = (__bf16)v;
}

// ---------------------------------------------------------------------------
// WMMA GEMM: C[M,N] = A * B^T + bias, A=[Mp x Kp] bf16, Bt=[Np x Kp] bf16,
// f32 accumulate. One wave per 16x16 tile, 4 waves/block, no guards in K loop.
// Fragment layouts used here (wave32, contiguous K, same order in A and B):
//   A 16x32: lanes 0-15 -> M=l15, elems K 0..15; lanes 16-31 K 16..31
//   B 32x16: lanes 0-15 -> N=l15, elems K 0..15; lanes 16-31 K 16..31
//   C/D: VGPR r -> M = 8*half + r, N = l15
// ---------------------------------------------------------------------------
__global__ void gemm_wmma_bf16(const __bf16* __restrict__ A,
                               const __bf16* __restrict__ Bt,
                               const float* __restrict__ bias,
                               float* __restrict__ C, int ldc,
                               int M, int N, int Kp, int ntiles) {
  int wid  = threadIdx.x >> 5;
  int tile = blockIdx.x * (blockDim.x >> 5) + wid;
  if (tile >= ntiles) return;
  int lane = threadIdx.x & 31;

  int ntn = (N + 15) >> 4;
  int tm  = (tile / ntn) << 4;
  int tn  = (tile % ntn) << 4;

  int hf  = lane >> 4;
  int l15 = lane & 15;

  const __bf16* ap = A  + (size_t)(tm + l15) * Kp + hf * 16;
  const __bf16* bp = Bt + (size_t)(tn + l15) * Kp + hf * 16;

  v8f acc = {};
#pragma unroll 2
  for (int kb = 0; kb < Kp; kb += 32) {
    v8bf  alo = *(const v8bf*)(ap + kb);        // K kb+hf*16 .. +7
    v8bf  ahi = *(const v8bf*)(ap + kb + 8);    // K kb+hf*16+8 .. +15
    v16bf a   = __builtin_shufflevector(alo, ahi, 0, 1, 2, 3, 4, 5, 6, 7,
                                        8, 9, 10, 11, 12, 13, 14, 15);
    v16bf b   = *(const v16bf*)(bp + kb);       // K kb+hf*16 .. +15
    acc = __builtin_amdgcn_wmma_f32_16x16x32_bf16(false, a, false, b,
                                                  (short)0, acc, false, false);
  }

  int cn = tn + l15;
  if (cn < N) {
    float bv = bias ? bias[cn] : 0.f;
#pragma unroll
    for (int r = 0; r < 8; ++r) {
      int cm = tm + hf * 8 + r;
      if (cm < M) C[(size_t)cm * ldc + cn] = acc[r] + bv;
    }
  }
}

// ---------------------------------------------------------------------------
// GCN aggregation: scatter h[src] into agg[dst] (+degree), then mean + tanh
// ---------------------------------------------------------------------------
__global__ void fill_f32(float* __restrict__ p, float v, int n) {
  int t = blockIdx.x * blockDim.x + threadIdx.x;
  if (t < n) p[t] = v;
}

__global__ void gcn_scatter(const float* __restrict__ h,
                            const int* __restrict__ src,
                            const int* __restrict__ dst,
                            float* __restrict__ agg,
                            float* __restrict__ cnt, int E) {
  long long t = (long long)blockIdx.x * blockDim.x + threadIdx.x;
  long long total = (long long)E * DIM;
  if (t >= total) return;
  int e = (int)(t >> 6);
  int d = (int)(t & 63);
  int s  = src[e];
  int dd = dst[e];
  atomicAdd(&agg[dd * DIM + d], h[s * DIM + d]);
  if (d == 0) atomicAdd(&cnt[dd], 1.0f);
}

__global__ void gcn_finalize(const float* __restrict__ agg,
                             const float* __restrict__ cnt,
                             float* __restrict__ out, int n) {
  int t = blockIdx.x * blockDim.x + threadIdx.x;
  if (t >= n) return;
  float c = cnt[t >> 6];
  c = c < 1.f ? 1.f : c;
  out[t] = tanhf(agg[t] / c);
}

// ---------------------------------------------------------------------------
// View fusion: c_u (item side, row L2-norm), c_i (user side, column L2-norm)
// ---------------------------------------------------------------------------
__global__ void fuse_item_rows(const float* __restrict__ s0, const float* __restrict__ s1,
                               const float* __restrict__ s2, const float* __restrict__ s3,
                               float* __restrict__ cu) {
  __shared__ float sh[DIM];
  int r = blockIdx.x;
  int d = threadIdx.x;
  const float* srcs[4] = {s0, s1, s2, s3};
  float out = 0.f;
  for (int s = 0; s < 4; ++s) {
    float v = srcs[s][r * DIM + d];
    sh[d] = v * v;
    __syncthreads();
    for (int off = 32; off > 0; off >>= 1) {
      if (d < off) sh[d] += sh[d + off];
      __syncthreads();
    }
    float sum = sh[0];
    __syncthreads();
    out += v * rsqrtf(sum);
  }
  cu[r * DIM + d] = 0.25f * out;
}

__global__ void user_colnorm(const float* __restrict__ s0, const float* __restrict__ s1,
                             const float* __restrict__ s2, const float* __restrict__ s3,
                             float* __restrict__ rn) {
  int t = blockIdx.x * blockDim.x + threadIdx.x;
  if (t >= 4 * DIM) return;
  int s = t >> 6, d = t & 63;
  const float* p = (s == 0) ? s0 : (s == 1) ? s1 : (s == 2) ? s2 : s3;
  float acc = 0.f;
  for (int r = 0; r < NUSER; ++r) { float v = p[r * DIM + d]; acc += v * v; }
  rn[t] = rsqrtf(acc);
}

__global__ void fuse_user_rows(const float* __restrict__ s0, const float* __restrict__ s1,
                               const float* __restrict__ s2, const float* __restrict__ s3,
                               const float* __restrict__ rn, float* __restrict__ ci) {
  int t = blockIdx.x * blockDim.x + threadIdx.x;
  if (t >= NUSER * DIM) return;
  int d = t & 63;
  float v = s0[t] * rn[d] + s1[t] * rn[DIM + d] +
            s2[t] * rn[2 * DIM + d] + s3[t] * rn[3 * DIM + d];
  ci[t] = 0.25f * v;
}

// ---------------------------------------------------------------------------
// Mean-pool normalization + BatchNorm(train) + ReLU
// ---------------------------------------------------------------------------
__global__ void row_sum(const float* __restrict__ presc, float* __restrict__ rs) {
  int b = blockIdx.x * blockDim.x + threadIdx.x;
  if (b >= BATCH) return;
  float s = 0.f;
  for (int k = 0; k < NITEM; ++k) s += presc[b * NITEM + k];
  rs[b] = s;
}

__global__ void scale_rows(float* __restrict__ e, const float* __restrict__ rs, int n) {
  int t = blockIdx.x * blockDim.x + threadIdx.x;
  if (t >= n) return;
  e[t] = e[t] / rs[t >> 6];
}

__global__ void bn_stats(const float* __restrict__ h, const float* __restrict__ gamma,
                         const float* __restrict__ beta, float* __restrict__ ab) {
  __shared__ float s1[256], s2[256];
  int d = blockIdx.x;
  int t = threadIdx.x;
  float a = 0.f, b = 0.f;
  for (int r = t; r < BATCH; r += 256) {
    float v = h[r * DIM + d];
    a += v; b += v * v;
  }
  s1[t] = a; s2[t] = b;
  __syncthreads();
  for (int off = 128; off > 0; off >>= 1) {
    if (t < off) { s1[t] += s1[t + off]; s2[t] += s2[t + off]; }
    __syncthreads();
  }
  if (t == 0) {
    float mu  = s1[0] / (float)BATCH;
    float var = s2[0] / (float)BATCH - mu * mu;   // biased, matches torch BN train
    float sc  = gamma[d] * rsqrtf(var + 1e-5f);
    ab[d]       = sc;
    ab[DIM + d] = beta[d] - mu * sc;
  }
}

__global__ void bn_apply_relu(float* __restrict__ h, const float* __restrict__ ab, int n) {
  int t = blockIdx.x * blockDim.x + threadIdx.x;
  if (t >= n) return;
  int d = t & 63;
  float v = h[t] * ab[d] + ab[DIM + d];
  h[t] = v > 0.f ? v : 0.f;
}

// ---------------------------------------------------------------------------
// Host orchestration
// ---------------------------------------------------------------------------
static void launch_gemm(const __bf16* A, const __bf16* Bt, const float* bias,
                        float* C, int ldc, int M, int N, int Kp, hipStream_t s) {
  int ntiles = ((M + 15) / 16) * ((N + 15) / 16);
  int wpb = 4;
  int blocks = (ntiles + wpb - 1) / wpb;
  gemm_wmma_bf16<<<blocks, 32 * wpb, 0, s>>>(A, Bt, bias, C, ldc, M, N, Kp, ntiles);
}

static void launch_pack_a(const float* src, int rows, int cols, __bf16* dst,
                          int prows, int pcols, hipStream_t s) {
  int n = prows * pcols;
  pack_a_bf16<<<(n + 255) / 256, 256, 0, s>>>(src, rows, cols, dst, pcols, n);
}

extern "C" void kernel_launch(void* const* d_in, const int* in_sizes, int n_in,
                              void* d_out, int out_size, void* d_ws, size_t ws_size,
                              hipStream_t stream) {
  const float* presc = (const float*)d_in[1];
  const float* emb   = (const float*)d_in[2];
  const float* W1    = (const float*)d_in[3];
  const float* b1    = (const float*)d_in[4];
  const float* W2    = (const float*)d_in[5];
  const float* b2    = (const float*)d_in[6];
  const float* mlpW  = (const float*)d_in[7];
  const float* mlpb  = (const float*)d_in[8];
  const float* gamma = (const float*)d_in[9];
  const float* beta  = (const float*)d_in[10];
  const int* gT = (const int*)d_in[11];
  const int* g1 = (const int*)d_in[12];
  const int* g2 = (const int*)d_in[13];
  int ET = in_sizes[11] / 2;
  int E1 = in_sizes[12] / 2;
  int E2 = in_sizes[13] / 2;

  const int ND  = NNODES * DIM;
  const int MPn = ceil16(NNODES);      // 1200, padded node-row count
  const int KPp = ceil32(NITEM);       // 416, padded prescription K
  const int NPu = ceil16(NUSER);       // 816, padded c_i rows

  // f32 scratch region
  float* ws = (float*)d_ws;
  size_t off = 0;
  float* xa    = ws + off; off += ND;            // linear output / layer scratch
  float* xb    = ws + off; off += ND;            // layer-1 activations
  float* agg   = ws + off; off += ND;
  float* cnt   = ws + off; off += NNODES;
  float* out6  = ws + off; off += ND;            // x_SH6
  float* out3  = ws + off; off += ND;            // SH_g3
  float* out33 = ws + off; off += ND;            // SH_g33
  float* cu    = ws + off; off += NITEM * DIM;
  float* ci    = ws + off; off += NUSER * DIM;
  float* rn    = ws + off; off += 4 * DIM;
  float* esynd = ws + off; off += BATCH * DIM;
  float* rs    = ws + off; off += BATCH;
  float* hbuf  = ws + off; off += BATCH * DIM;
  float* bnab  = ws + off; off += 2 * DIM;
  off = (off + 15) & ~(size_t)15;                // 64B-align bf16 region

  // bf16 scratch region (reused sequentially per GEMM)
  __bf16* aPack = (__bf16*)(ws + off);           // up to 2048 x 416
  __bf16* bPack = aPack + (size_t)BATCH * KPp;   // up to 816 x 64

  auto gcn_layer = [&](const float* xin, const float* W, const float* b,
                       const int* src, const int* dst, int E, float* xout) {
    // h = xin @ W^T + b  (W is [out][in] = [N][K] row-major already)
    launch_pack_a(xin, NNODES, DIM, aPack, MPn, DIM, stream);
    launch_pack_a(W, DIM, DIM, bPack, DIM, DIM, stream);
    launch_gemm(aPack, bPack, b, xa, DIM, NNODES, DIM, DIM, stream);
    // segment-mean over edges + tanh
    fill_f32<<<(ND + 255) / 256, 256, 0, stream>>>(agg, 0.f, ND);
    fill_f32<<<(NNODES + 255) / 256, 256, 0, stream>>>(cnt, 0.f, NNODES);
    long long tot = (long long)E * DIM;
    gcn_scatter<<<(int)((tot + 255) / 256), 256, 0, stream>>>(xa, src, dst, agg, cnt, E);
    gcn_finalize<<<(ND + 255) / 256, 256, 0, stream>>>(agg, cnt, xout, ND);
  };
  auto two_layer = [&](const int* g, int E, float* out) {
    gcn_layer(emb, W1, b1, g, g + E, E, xb);
    gcn_layer(xb, W2, b2, g, g + E, E, out);
  };

  two_layer(gT, ET, out6);   // x_SH6
  two_layer(g1, E1, out3);   // SH_g3
  two_layer(g2, E2, out33);  // SH_g33

  // c_u: item slices, row-normalized, averaged
  fuse_item_rows<<<NITEM, DIM, 0, stream>>>(emb + NUSER * DIM, out3 + NUSER * DIM,
                                            out6 + NUSER * DIM, out33 + NUSER * DIM, cu);
  // c_i: user slices, column-normalized, averaged
  user_colnorm<<<1, 256, 0, stream>>>(emb, out3, out6, out33, rn);
  fuse_user_rows<<<(NUSER * DIM + 255) / 256, 256, 0, stream>>>(emb, out3, out6, out33, rn, ci);

  // e_synd = (presc @ c_u) / rowsum(presc)
  row_sum<<<(BATCH + 255) / 256, 256, 0, stream>>>(presc, rs);
  launch_pack_a(presc, BATCH, NITEM, aPack, BATCH, KPp, stream);
  {
    int n = DIM * KPp;   // Bt = c_u^T : [64 x 416]
    pack_bt_bf16<<<(n + 255) / 256, 256, 0, stream>>>(cu, NITEM, DIM, bPack, KPp, n);
  }
  launch_gemm(aPack, bPack, nullptr, esynd, DIM, BATCH, DIM, KPp, stream);
  scale_rows<<<(BATCH * DIM + 255) / 256, 256, 0, stream>>>(esynd, rs, BATCH * DIM);

  // h = e_synd @ mlp_W^T + mlp_b, then BN(train) + ReLU
  launch_pack_a(esynd, BATCH, DIM, aPack, BATCH, DIM, stream);
  launch_pack_a(mlpW, DIM, DIM, bPack, DIM, DIM, stream);
  launch_gemm(aPack, bPack, mlpb, hbuf, DIM, BATCH, DIM, DIM, stream);
  bn_stats<<<DIM, 256, 0, stream>>>(hbuf, gamma, beta, bnab);
  bn_apply_relu<<<(BATCH * DIM + 255) / 256, 256, 0, stream>>>(hbuf, bnab, BATCH * DIM);

  // pre = h @ c_i^T -> [2048, 805]
  launch_pack_a(hbuf, BATCH, DIM, aPack, BATCH, DIM, stream);
  launch_pack_a(ci, NUSER, DIM, bPack, NPu, DIM, stream);
  launch_gemm(aPack, bPack, nullptr, (float*)d_out, NUSER, BATCH, NUSER, DIM, stream);
}
